// MambaSemanticMapper_NoPool_50208167690330
// MI455X (gfx1250) — matrix operand order repaired and checked by reading, not compiled, not run on hardware
//
#include <hip/hip_runtime.h>
#include <hip/hip_bf16.h>

#define LSEQ 1024
#define BB 4
#define BL (BB * LSEQ)          // 4096 rows
#define DMODEL 256
#define DCLIP 512
#define DINNER 512
#define DSTATE 16
#define DTRANK 16
#define NPROJ 48                // real proj cols
#define NPROJP 64               // padded to multiple of 32 for 32x32 GEMM tiles

typedef _Float16 v8h  __attribute__((ext_vector_type(8)));
typedef _Float16 v16h __attribute__((ext_vector_type(16)));
typedef float    v8f  __attribute__((ext_vector_type(8)));

union FragH { v16h v; v8h h[2]; };

enum { GF_BIAS = 1, GF_WRITEH = 2, GF_REVA = 4, GF_REVC = 8, GF_ACC = 16 };

// ---------------------------------------------------------------------------
// fp32 -> fp16 convert
// ---------------------------------------------------------------------------
__global__ void k_f32_to_f16(const float* __restrict__ in, _Float16* __restrict__ out, int n) {
    int i = blockIdx.x * blockDim.x + threadIdx.x;
    if (i < n) out[i] = (_Float16)in[i];
}

// fp32 -> fp16 with zero-padded extra rows (for Wx: 48x512 -> 64x512)
__global__ void k_f32_to_f16_pad(const float* __restrict__ in, _Float16* __restrict__ out,
                                 int rows_in, int rows_out, int cols) {
    int i = blockIdx.x * blockDim.x + threadIdx.x;
    if (i >= rows_out * cols) return;
    int row = i / cols;
    out[i] = (row < rows_in) ? (_Float16)in[i] : (_Float16)0.0f;
}

// ---------------------------------------------------------------------------
// TN GEMM via v_wmma_f32_16x16x32_f16:  C[M,N] = A[M,K] @ W[N,K]^T (+bias)
// One wave32 per 32x32 output tile (2x2 register tiling: 4 WMMAs per K-step
// from 8 b128 loads -> 2 loads/WMMA). M,N multiples of 32, K multiple of 32.
// REV_A: read A rows time-reversed within each length-LSEQ segment.
// REV_C: write/accumulate C rows time-reversed. ACC: C += instead of C =.
// ---------------------------------------------------------------------------
__global__ void k_gemm_f16(const _Float16* __restrict__ A, const _Float16* __restrict__ W,
                           const float* __restrict__ bias, float* __restrict__ C,
                           _Float16* __restrict__ Ch, int M, int N, int K, int flags) {
    const int wave = blockIdx.x * (blockDim.x >> 5) + (threadIdx.x >> 5);
    const int lane = threadIdx.x & 31;
    const int ntiles = N >> 5;
    const int mt = wave / ntiles, nt = wave % ntiles;
    const int row0 = mt << 5, col0 = nt << 5;
    const int half = lane >> 4;
    const int lidx = lane & 15;

    // A-fragment rows (16-bit A 16x32: lane m holds K[8h..8h+7] and K[8h+16..8h+23])
    int am0 = row0 + lidx;
    int am1 = row0 + 16 + lidx;
    if (flags & GF_REVA) {
        am0 = (am0 / LSEQ) * LSEQ + (LSEQ - 1 - (am0 % LSEQ));
        am1 = (am1 / LSEQ) * LSEQ + (LSEQ - 1 - (am1 % LSEQ));
    }
    const _Float16* __restrict__ arow0 = A + (size_t)am0 * K + half * 8;
    const _Float16* __restrict__ arow1 = A + (size_t)am1 * K + half * 8;
    // B-fragment: lane n holds K[8h..8h+7] and K[8h+16..8h+23] from row n of W
    const _Float16* __restrict__ wrow0 = W + (size_t)(col0 + lidx) * K + half * 8;
    const _Float16* __restrict__ wrow1 = W + (size_t)(col0 + 16 + lidx) * K + half * 8;

    v8f acc00 = {}, acc01 = {}, acc10 = {}, acc11 = {};
#pragma unroll 2
    for (int k0 = 0; k0 < K; k0 += 32) {
        FragH a0, a1, b0, b1;
        a0.h[0] = *(const v8h*)(arow0 + k0);
        a0.h[1] = *(const v8h*)(arow0 + k0 + 16);
        a1.h[0] = *(const v8h*)(arow1 + k0);
        a1.h[1] = *(const v8h*)(arow1 + k0 + 16);
        b0.h[0] = *(const v8h*)(wrow0 + k0);
        b0.h[1] = *(const v8h*)(wrow0 + k0 + 16);
        b1.h[0] = *(const v8h*)(wrow1 + k0);
        b1.h[1] = *(const v8h*)(wrow1 + k0 + 16);
        acc00 = __builtin_amdgcn_wmma_f32_16x16x32_f16(false, a0.v, false, b0.v,
                                                       (short)0, acc00, false, false);
        acc01 = __builtin_amdgcn_wmma_f32_16x16x32_f16(false, a0.v, false, b1.v,
                                                       (short)0, acc01, false, false);
        acc10 = __builtin_amdgcn_wmma_f32_16x16x32_f16(false, a1.v, false, b0.v,
                                                       (short)0, acc10, false, false);
        acc11 = __builtin_amdgcn_wmma_f32_16x16x32_f16(false, a1.v, false, b1.v,
                                                       (short)0, acc11, false, false);
    }

    const v8f* accs[4] = { &acc00, &acc01, &acc10, &acc11 };
#pragma unroll
    for (int t = 0; t < 4; t++) {
        const int cy = t >> 1, cx = t & 1;
        const v8f av = *accs[t];
#pragma unroll
        for (int r = 0; r < 8; r++) {
            const int m   = cy * 16 + r + half * 8;  // C/D: VGPR r holds M=r (+8 hi lanes)
            const int row = row0 + m;
            const int col = col0 + cx * 16 + lidx;
            float v = av[r];
            if (flags & GF_BIAS) v += bias[col];
            int orow = row;
            if (flags & GF_REVC) orow = (row / LSEQ) * LSEQ + (LSEQ - 1 - (row % LSEQ));
            const size_t oi = (size_t)orow * N + col;
            if (flags & GF_ACC) C[oi] += v; else C[oi] = v;
            if (flags & GF_WRITEH) Ch[oi] = (_Float16)v;
        }
    }
}

// ---------------------------------------------------------------------------
// Depthwise causal conv (k=4) + bias + SiLU.  xin = cols [0,512) of xz[BL,1024].
// ---------------------------------------------------------------------------
__global__ void k_conv_silu(const float* __restrict__ xz, const float* __restrict__ convw,
                            const float* __restrict__ convb, float* __restrict__ xc,
                            _Float16* __restrict__ xch) {
    const int idx = blockIdx.x * blockDim.x + threadIdx.x;   // BL*DINNER threads
    const int d = idx & (DINNER - 1);
    const int r = idx >> 9;
    const int l = r & (LSEQ - 1);
    const int b = r >> 10;
    float v = convb[d];
#pragma unroll
    for (int j = 0; j < 4; j++) {
        const int li = l - 3 + j;
        if (li >= 0) v += convw[d * 4 + j] * xz[((size_t)(b * LSEQ + li)) * (2 * DINNER) + d];
    }
    const float s = v / (1.0f + __expf(-v));
    xc[idx] = s;
    xch[idx] = (_Float16)s;
}

// ---------------------------------------------------------------------------
// dt = softplus(dt_r @ Wdt^T + bdt)   (K=16: VALU). proj has stride NPROJP.
// ---------------------------------------------------------------------------
__global__ void k_dt(const float* __restrict__ proj, const float* __restrict__ Wdt,
                     const float* __restrict__ bdt, float* __restrict__ dt) {
    const int idx = blockIdx.x * blockDim.x + threadIdx.x;   // BL*DINNER threads
    const int d = idx & (DINNER - 1);
    const int r = idx >> 9;
    const float* __restrict__ p = proj + (size_t)r * NPROJP;
    float a = bdt[d];
#pragma unroll
    for (int j = 0; j < DTRANK; j++) a += p[j] * Wdt[d * DTRANK + j];
    dt[idx] = (a > 20.0f) ? a : log1pf(__expf(a));
}

// ---------------------------------------------------------------------------
// Selective scan: one lane per (b, d, s) -> 32768 lanes, 1024 wave32s.
// 16-lane shfl-xor reduction over s for y[l,d] = sum_s h*C + D*xc.
// ---------------------------------------------------------------------------
__global__ void k_scan(const float* __restrict__ dt, const float* __restrict__ xc,
                       const float* __restrict__ proj, const float* __restrict__ Alog,
                       const float* __restrict__ Dp, float* __restrict__ y) {
    const int g = blockIdx.x * (blockDim.x >> 4) + (threadIdx.x >> 4);  // (b,d) group
    const int s = threadIdx.x & 15;
    const int d = g & (DINNER - 1);
    const int b = g >> 9;
    const float As = -__expf(Alog[d * DSTATE + s]);
    const float Dd = Dp[d];
    float h = 0.0f;
    const size_t rbase = (size_t)b * LSEQ;
    for (int l = 0; l < LSEQ; l++) {
        const size_t r = rbase + l;
        const float dtv = dt[r * DINNER + d];
        const float xcv = xc[r * DINNER + d];
        const float Bv  = proj[r * NPROJP + DTRANK + s];
        const float Cv  = proj[r * NPROJP + DTRANK + DSTATE + s];
        h = __expf(dtv * As) * h + dtv * Bv * xcv;
        float c = h * Cv;
        c += __shfl_xor(c, 1, 16);
        c += __shfl_xor(c, 2, 16);
        c += __shfl_xor(c, 4, 16);
        c += __shfl_xor(c, 8, 16);
        if (s == 0) y[r * DINNER + d] = c + Dd * xcv;
    }
}

// ---------------------------------------------------------------------------
// Gate: yg = y * silu(z), z = cols [512,1024) of xz. Emit f16 for Wout GEMM.
// ---------------------------------------------------------------------------
__global__ void k_gate(const float* __restrict__ xz, const float* __restrict__ y,
                       _Float16* __restrict__ ygh) {
    const int idx = blockIdx.x * blockDim.x + threadIdx.x;   // BL*DINNER threads
    const int d = idx & (DINNER - 1);
    const int r = idx >> 9;
    const float zv = xz[(size_t)r * (2 * DINNER) + DINNER + d];
    const float gl = zv / (1.0f + __expf(-zv));
    ygh[idx] = (_Float16)(y[idx] * gl);
}

// ---------------------------------------------------------------------------
// Host launch
// ---------------------------------------------------------------------------
extern "C" void kernel_launch(void* const* d_in, const int* in_sizes, int n_in,
                              void* d_out, int out_size, void* d_ws, size_t ws_size,
                              hipStream_t stream) {
    (void)in_sizes; (void)n_in; (void)out_size; (void)ws_size;
    const float* ts = (const float*)d_in[0];
    const float* Wp = (const float*)d_in[1];
    const float* bp = (const float*)d_in[2];
    const float *Win[2], *convw[2], *convb[2], *Wx[2], *Wdt[2], *bdt[2], *Alog[2], *Dp[2], *Wout[2];
    for (int dir = 0; dir < 2; dir++) {
        const int o = 3 + dir * 9;
        Win[dir]   = (const float*)d_in[o + 0];
        convw[dir] = (const float*)d_in[o + 1];
        convb[dir] = (const float*)d_in[o + 2];
        Wx[dir]    = (const float*)d_in[o + 3];
        Wdt[dir]   = (const float*)d_in[o + 4];
        bdt[dir]   = (const float*)d_in[o + 5];
        Alog[dir]  = (const float*)d_in[o + 6];
        Dp[dir]    = (const float*)d_in[o + 7];
        Wout[dir]  = (const float*)d_in[o + 8];
    }
    float* out = (float*)d_out;

    // bump allocator over workspace
    char* w = (char*)d_ws;
    size_t off = 0;
    auto alloc = [&](size_t bytes) -> char* {
        char* p = w + off;
        off = (off + bytes + 255) & ~(size_t)255;
        return p;
    };
    _Float16* tsh      = (_Float16*)alloc((size_t)BL * DCLIP * 2);
    _Float16* Wph      = (_Float16*)alloc((size_t)DMODEL * DCLIP * 2);
    _Float16* Winh[2]  = { (_Float16*)alloc((size_t)2 * DINNER * DMODEL * 2),
                           (_Float16*)alloc((size_t)2 * DINNER * DMODEL * 2) };
    _Float16* Wxh[2]   = { (_Float16*)alloc((size_t)NPROJP * DINNER * 2),
                           (_Float16*)alloc((size_t)NPROJP * DINNER * 2) };
    _Float16* Wouth[2] = { (_Float16*)alloc((size_t)DMODEL * DINNER * 2),
                           (_Float16*)alloc((size_t)DMODEL * DINNER * 2) };
    float*    x    = (float*)alloc((size_t)BL * DMODEL * 4);
    _Float16* xh   = (_Float16*)alloc((size_t)BL * DMODEL * 2);
    float*    xz   = (float*)alloc((size_t)BL * 2 * DINNER * 4);
    float*    xc   = (float*)alloc((size_t)BL * DINNER * 4);
    _Float16* xch  = (_Float16*)alloc((size_t)BL * DINNER * 2);
    float*    proj = (float*)alloc((size_t)BL * NPROJP * 4);
    float*    dtb  = (float*)alloc((size_t)BL * DINNER * 4);
    float*    yb   = (float*)alloc((size_t)BL * DINNER * 4);
    _Float16* ygh  = (_Float16*)alloc((size_t)BL * DINNER * 2);

    auto cvt = [&](const float* src, _Float16* dst, int n) {
        k_f32_to_f16<<<(n + 255) / 256, 256, 0, stream>>>(src, dst, n);
    };
    cvt(ts, tsh, BL * DCLIP);
    cvt(Wp, Wph, DMODEL * DCLIP);
    for (int dir = 0; dir < 2; dir++) {
        cvt(Win[dir], Winh[dir], 2 * DINNER * DMODEL);
        k_f32_to_f16_pad<<<(NPROJP * DINNER + 255) / 256, 256, 0, stream>>>(
            Wx[dir], Wxh[dir], NPROJ, NPROJP, DINNER);
        cvt(Wout[dir], Wouth[dir], DMODEL * DINNER);
    }

    const int NEL = BL * DINNER;                 // 2,097,152
    const int ELB = NEL / 256;                   // 8192 blocks for elementwise kernels

    // 32x32 tiles, 4 waves (128 threads) per block
    auto gemm_blocks = [](int M, int N) { return (M / 32) * (N / 32) / 4; };

    // x = ts @ Wp^T + bp  (M=4096, N=256, K=512); also emit f16 shadow
    k_gemm_f16<<<gemm_blocks(BL, DMODEL), 128, 0, stream>>>(
        tsh, Wph, bp, x, xh, BL, DMODEL, DCLIP, GF_BIAS | GF_WRITEH);

    for (int dir = 0; dir < 2; dir++) {
        // xz = x(rev?) @ Win^T  (M=4096, N=1024, K=256)
        k_gemm_f16<<<gemm_blocks(BL, 2 * DINNER), 128, 0, stream>>>(
            xh, Winh[dir], nullptr, xz, nullptr, BL, 2 * DINNER, DMODEL,
            dir ? GF_REVA : 0);
        // depthwise conv + SiLU
        k_conv_silu<<<ELB, 256, 0, stream>>>(xz, convw[dir], convb[dir], xc, xch);
        // proj = xc @ Wx^T  (M=4096, N=64 padded, K=512)
        k_gemm_f16<<<gemm_blocks(BL, NPROJP), 128, 0, stream>>>(
            xch, Wxh[dir], nullptr, proj, nullptr, BL, NPROJP, DINNER, 0);
        // dt = softplus(dt_r @ Wdt^T + bdt)
        k_dt<<<ELB, 256, 0, stream>>>(proj, Wdt[dir], bdt[dir], dtb);
        // selective scan: 32768 lanes
        k_scan<<<(BB * DINNER) / 16, 256, 0, stream>>>(dtb, xc, proj, Alog[dir], Dp[dir], yb);
        // gate with silu(z)
        k_gate<<<ELB, 256, 0, stream>>>(xz, yb, ygh);
        // out (+)= yg @ Wout^T  (M=4096, N=256, K=512); reverse rows + accumulate for dir 1
        k_gemm_f16<<<gemm_blocks(BL, DMODEL), 128, 0, stream>>>(
            ygh, Wouth[dir], nullptr, out, nullptr, BL, DMODEL, DINNER,
            dir ? (GF_REVC | GF_ACC) : 0);
    }
}
